// DenseCoAttn_42150809043407
// MI455X (gfx1250) — matrix operand order repaired and checked by reading, not compiled, not run on hardware
//
#include <hip/hip_runtime.h>

typedef __bf16 bf16_t;
typedef __attribute__((ext_vector_type(16))) __bf16 v16bf;
typedef __attribute__((ext_vector_type(8)))  __bf16 v8bf;
typedef __attribute__((ext_vector_type(8)))  float  v8f;

#define WMMA_BF16(a, b, c) \
  __builtin_amdgcn_wmma_f32_16x16x32_bf16(false, (a), false, (b), (short)0, (c), false, false)

// Problem constants
constexpr int Bn  = 128;
constexpr int L1  = 196;
constexpr int L2  = 30;
constexpr int D1  = 2048;
constexpr int D2  = 1024;
constexpr int DIM = 1024;
constexpr int H   = 8;
constexpr int DK  = 128;
constexpr int L1P = 224;   // 199 padded to multiple of 32
constexpr int L2P = 64;    // 33  padded to multiple of 32

// ---------- CDNA5 async copy (global -> LDS, ASYNCcnt-tracked) ----------

// Per-lane: LDS[lds_off] = MEM[gaddr], 16 bytes. Tracked by ASYNCcnt.
__device__ inline void async_copy_b128(unsigned lds_off, const void* gaddr) {
  asm volatile("global_load_async_to_lds_b128 %0, %1, off"
               :: "v"(lds_off), "v"(gaddr)
               : "memory");
}
__device__ inline void wait_async0() {
  asm volatile("s_wait_asynccnt 0x0" ::: "memory");
}
// Low 32 bits of a generic pointer into LDS == LDS byte offset (ISA 10.2 aperture rule).
__device__ inline unsigned lds_off32(const void* p) {
  return (unsigned)(unsigned long long)p;
}

// ---------- fragment helpers (CDNA5 16-bit WMMA layouts, wave32) ----------

// A 16x32 bf16 fragment: lane<16 -> row M=lane, K = k0..k0+15
//                        lane>=16 -> row M=lane-16, K = k0+16..k0+31
__device__ inline v16bf a_frag_bf16(const bf16_t* row, int k0, int hi) {
  const int kb = k0 + (hi ? 16 : 0);
  v8bf lo = *(const v8bf*)(row + kb);
  v8bf hh = *(const v8bf*)(row + kb + 8);
  v16bf r;
#pragma unroll
  for (int i = 0; i < 8; ++i) { r[i] = lo[i]; r[i + 8] = hh[i]; }
  return r;
}

// B 32x16 bf16 fragment: lane n<16 holds B[k0+0..15][n], lane n+16 holds B[k0+16..31][n].
// Source is stored K-contiguous per column n ("row" points at column n's K-run).
__device__ inline v16bf b_frag(const bf16_t* __restrict__ row, int k0, int hi) {
  return *(const v16bf*)(row + k0 + (hi ? 16 : 0));
}

__device__ inline float wave_max(float v) {
#pragma unroll
  for (int m = 16; m >= 1; m >>= 1) v = fmaxf(v, __shfl_xor(v, m, 32));
  return v;
}
__device__ inline float wave_sum(float v) {
#pragma unroll
  for (int m = 16; m >= 1; m >>= 1) v += __shfl_xor(v, m, 32);
  return v;
}

// Padded value row r: rows 0..2 = none tokens, 3..3+L-1 = value, else zero (nullptr).
__device__ inline const float* vrow(const float* __restrict__ value,
                                    const float* __restrict__ none,
                                    int b, int r, int L, int D) {
  if (r < 3) return none + (size_t)r * D;
  if (r < 3 + L) return value + ((size_t)b * L + (size_t)(r - 3)) * D;
  return nullptr;
}

// ---------- kernels ----------

__global__ void cvt_f32_bf16(const float* __restrict__ in, bf16_t* __restrict__ out, int n) {
  int i = blockIdx.x * blockDim.x + threadIdx.x;
  if (i < n) out[i] = (bf16_t)in[i];
}

// Build vT[b][d][k] = bf16(padded_v[b][k][d]); LDS-tiled 32x32 transpose.
__global__ void pack_transpose(const float* __restrict__ value, const float* __restrict__ none,
                               bf16_t* __restrict__ vT, int L, int D, int Kp) {
  __shared__ float t[32][33];
  const int b = blockIdx.x, k0 = blockIdx.y * 32, d0 = blockIdx.z * 32;
  const int tx = threadIdx.x, ty = threadIdx.y;
#pragma unroll
  for (int i = 0; i < 4; ++i) {
    const int kk = k0 + ty + i * 8;
    const int dd = d0 + tx;
    float v = 0.0f;
    if (kk < 3)            v = none[(size_t)kk * D + dd];
    else if (kk < 3 + L)   v = value[((size_t)b * L + (size_t)(kk - 3)) * D + dd];
    t[ty + i * 8][tx] = v;
  }
  __syncthreads();
#pragma unroll
  for (int i = 0; i < 4; ++i) {
    const int dd = d0 + ty + i * 8;
    vT[((size_t)b * D + dd) * Kp + k0 + tx] = (bf16_t)t[tx][ty + i * 8];
  }
}

// Q = padded_v @ W^T. A-tile (16 x D) staged in LDS as bf16 once per block;
// each wave computes a 16x32 output tile (A-fragment reused for 2 WMMAs).
__global__ __launch_bounds__(256) void proj_gemm(const float* __restrict__ value,
                                                 const float* __restrict__ none,
                                                 const bf16_t* __restrict__ Wb,
                                                 bf16_t* __restrict__ Qout,
                                                 int L, int D, int Mp) {
  __shared__ bf16_t atile[16 * D1];  // 64 KB max (D1=2048); D2 kernel uses half
  const int b = blockIdx.x;
  const int m0 = blockIdx.y * 16;
  const int tid = threadIdx.x, wave = tid >> 5, lane = tid & 31, ln = lane & 15, hi = lane >> 4;

  // Cooperative stage: 16 padded rows -> bf16 in LDS (zeros for pad rows).
  {
    const int c0 = tid * 8;
    if (c0 < D) {
#pragma unroll 1
      for (int r = 0; r < 16; ++r) {
        const float* src = vrow(value, none, b, m0 + r, L, D);
        v8bf o;
        if (src) {
#pragma unroll
          for (int j = 0; j < 8; ++j) o[j] = (bf16_t)src[c0 + j];
        } else {
#pragma unroll
          for (int j = 0; j < 8; ++j) o[j] = (bf16_t)0.0f;
        }
        *(v8bf*)(atile + r * D + c0) = o;
      }
    }
  }
  __syncthreads();

  const int n0 = blockIdx.z * 256 + wave * 32;
  const bf16_t* brow0 = Wb + (size_t)(n0 + ln) * D;
  const bf16_t* brow1 = Wb + (size_t)(n0 + 16 + ln) * D;
  const bf16_t* alds = atile + ln * D;
  v8f acc0 = {}, acc1 = {};
  for (int k0 = 0; k0 < D; k0 += 32) {
    __builtin_prefetch(brow0 + k0 + 256, 0, 1);
    __builtin_prefetch(brow1 + k0 + 256, 0, 1);
    v16bf a = a_frag_bf16(alds, k0, hi);
    acc0 = WMMA_BF16(a, b_frag(brow0, k0, hi), acc0);
    acc1 = WMMA_BF16(a, b_frag(brow1, k0, hi), acc1);
  }
  bf16_t* q = Qout + (size_t)b * Mp * DIM;
#pragma unroll
  for (int rr = 0; rr < 8; ++rr) {
    const int m = m0 + hi * 8 + rr;
    q[(size_t)m * DIM + n0 + ln]      = (bf16_t)acc0[rr];
    q[(size_t)m * DIM + n0 + 16 + ln] = (bf16_t)acc1[rr];
  }
}

// Fused: scores (WMMA) -> masked scaled softmax -> mean over 8 heads. 16 Q rows per block.
// The 16x1024 Q tile is async-copied to LDS once and reused by all heads/n-tiles.
__global__ __launch_bounds__(256) void attn_probs(const bf16_t* __restrict__ Q,
                                                  const bf16_t* __restrict__ K,
                                                  const float* __restrict__ mask,
                                                  bf16_t* __restrict__ P,
                                                  int Np, int Mp, int L) {
  __shared__ float sc[16 * L1P];
  __shared__ float pacc[16 * L1P];
  __shared__ float mval[L1P];
  __shared__ bf16_t qtile[16 * DIM];  // 32 KB
  const int b = blockIdx.x;
  const int m0 = blockIdx.y * 16;
  const int tid = threadIdx.x, wave = tid >> 5, lane = tid & 31, ln = lane & 15, hi = lane >> 4;

  // Async-stage Q tile (rows m0..m0+15, full 1024-wide) into LDS.
  const bf16_t* Qb = Q + (size_t)b * Mp * DIM + (size_t)m0 * DIM;
  for (int i = tid * 8; i < 16 * DIM; i += 256 * 8)
    async_copy_b128(lds_off32(qtile + i), Qb + i);

  for (int c = tid; c < Np; c += 256)
    mval[c] = (c < 3) ? 1.0f : ((c < 3 + L) ? mask[(size_t)b * L + (c - 3)] : 0.0f);
  for (int i = tid; i < 16 * Np; i += 256) pacc[i] = 0.0f;
  wait_async0();
  __syncthreads();

  const bf16_t* Kb = K + (size_t)b * Np * DIM;
  const int NT = Np >> 4;
  const float scale = 0.08838834764831845f;  // 1/sqrt(128)

  for (int h = 0; h < H; ++h) {
    const int koff = h * DK;
    for (int nt = wave; nt < NT; nt += 8) {
      const bf16_t* arow = qtile + ln * DIM + koff;
      const bf16_t* brow = Kb + (size_t)(nt * 16 + ln) * DIM + koff;
      v8f acc = {};
#pragma unroll
      for (int k0 = 0; k0 < DK; k0 += 32)
        acc = WMMA_BF16(a_frag_bf16(arow, k0, hi), b_frag(brow, k0, hi), acc);
      const int col = nt * 16 + ln;
#pragma unroll
      for (int rr = 0; rr < 8; ++rr) sc[(hi * 8 + rr) * Np + col] = acc[rr];
    }
    __syncthreads();
    for (int row = wave; row < 16; row += 8) {
      float* s = sc + row * Np;
      float mx = -1e30f;
      for (int c = lane; c < Np; c += 32) {
        float v = (mval[c] != 0.0f) ? s[c] * scale : -1e9f;
        s[c] = v;
        mx = fmaxf(mx, v);
      }
      mx = wave_max(mx);
      float sum = 0.0f;
      for (int c = lane; c < Np; c += 32) {
        float e = __expf(s[c] - mx);
        s[c] = e;
        sum += e;
      }
      sum = wave_sum(sum);
      const float inv = 0.125f / sum;  // 1/H folded into the probabilities
      for (int c = lane; c < Np; c += 32) pacc[row * Np + c] += s[c] * inv;
    }
    __syncthreads();
  }
  bf16_t* Pb = P + (size_t)b * Mp * Np + (size_t)m0 * Np;
  for (int i = tid; i < 16 * Np; i += 256) Pb[i] = (bf16_t)pacc[i];
}

// out = Pavg @ v (via vT), f32 output, dropping the 3 none rows.
// All 8 waves share the same m-tile, so the 16 x Kp A-tile is async-staged in LDS
// once per block; each wave computes a 16x32 tile (A-fragment reused for 2 WMMAs).
__global__ __launch_bounds__(256) void pv_gemm(const bf16_t* __restrict__ P,
                                               const bf16_t* __restrict__ vT,
                                               float* __restrict__ out,
                                               int Kp, int N, int Lout, int Mp) {
  __shared__ bf16_t atile[16 * L1P];  // 7 KB max
  const int b = blockIdx.x, m0 = blockIdx.y * 16;
  const int tid = threadIdx.x, wave = tid >> 5, lane = tid & 31, ln = lane & 15, hi = lane >> 4;

  const bf16_t* Pb = P + (size_t)b * Mp * Kp + (size_t)m0 * Kp;
  for (int i = tid * 8; i < 16 * Kp; i += 256 * 8)
    async_copy_b128(lds_off32(atile + i), Pb + i);
  wait_async0();
  __syncthreads();

  const int n0 = blockIdx.z * 256 + wave * 32;
  const bf16_t* brow0 = vT + (size_t)b * N * Kp + (size_t)(n0 + ln) * Kp;
  const bf16_t* brow1 = vT + (size_t)b * N * Kp + (size_t)(n0 + 16 + ln) * Kp;
  const bf16_t* arow = atile + ln * Kp;
  v8f acc0 = {}, acc1 = {};
  for (int k0 = 0; k0 < Kp; k0 += 32) {
    v16bf a = a_frag_bf16(arow, k0, hi);
    acc0 = WMMA_BF16(a, b_frag(brow0, k0, hi), acc0);
    acc1 = WMMA_BF16(a, b_frag(brow1, k0, hi), acc1);
  }
#pragma unroll
  for (int rr = 0; rr < 8; ++rr) {
    const int m = m0 + hi * 8 + rr;
    if (m >= 3 && m < 3 + Lout) {
      float* o = out + (size_t)b * Lout * N + (size_t)(m - 3) * N;
      o[n0 + ln]      = acc0[rr];
      o[n0 + 16 + ln] = acc1[rr];
    }
  }
}

// ---------- host ----------

extern "C" void kernel_launch(void* const* d_in, const int* in_sizes, int n_in,
                              void* d_out, int out_size, void* d_ws, size_t ws_size,
                              hipStream_t stream) {
  const float* value1 = (const float*)d_in[0];
  const float* value2 = (const float*)d_in[1];
  const float* mask1  = (const float*)d_in[2];
  const float* mask2  = (const float*)d_in[3];
  const float* W1     = (const float*)d_in[4];
  const float* W2     = (const float*)d_in[5];
  const float* none1  = (const float*)d_in[6];
  const float* none2  = (const float*)d_in[7];

  char* ws = (char*)d_ws;
  size_t off = 0;
  bf16_t* W1b = (bf16_t*)(ws + off); off += (size_t)DIM * D1 * 2;        // 4 MB
  bf16_t* W2b = (bf16_t*)(ws + off); off += (size_t)DIM * D2 * 2;        // 2 MB
  bf16_t* v1T = (bf16_t*)(ws + off); off += (size_t)Bn * D1 * L1P * 2;   // 112 MB
  bf16_t* v2T = (bf16_t*)(ws + off); off += (size_t)Bn * D2 * L2P * 2;   // 16 MB
  bf16_t* q1b = (bf16_t*)(ws + off); off += (size_t)Bn * L1P * DIM * 2;  // 56 MB
  bf16_t* q2b = (bf16_t*)(ws + off); off += (size_t)Bn * L2P * DIM * 2;  // 16 MB
  bf16_t* P1  = (bf16_t*)(ws + off); off += (size_t)Bn * L2P * L1P * 2;  // 3.5 MB
  bf16_t* P2  = (bf16_t*)(ws + off); off += (size_t)Bn * L1P * L2P * 2;  // 3.5 MB

  float* out1 = (float*)d_out;                       // [B, L2, D1]
  float* out2 = out1 + (size_t)Bn * L2 * D1;         // [B, L1, D2]

  cvt_f32_bf16<<<dim3((DIM * D1 + 255) / 256), 256, 0, stream>>>(W1, W1b, DIM * D1);
  cvt_f32_bf16<<<dim3((DIM * D2 + 255) / 256), 256, 0, stream>>>(W2, W2b, DIM * D2);

  pack_transpose<<<dim3(Bn, L1P / 32, D1 / 32), dim3(32, 8), 0, stream>>>(value1, none1, v1T, L1, D1, L1P);
  pack_transpose<<<dim3(Bn, L2P / 32, D2 / 32), dim3(32, 8), 0, stream>>>(value2, none2, v2T, L2, D2, L2P);

  proj_gemm<<<dim3(Bn, L1P / 16, 4), 256, 0, stream>>>(value1, none1, W1b, q1b, L1, D1, L1P);
  proj_gemm<<<dim3(Bn, L2P / 16, 4), 256, 0, stream>>>(value2, none2, W2b, q2b, L2, D2, L2P);

  // weighted1: Q=q2 (heads), K=q1, mask over n+L1 columns
  attn_probs<<<dim3(Bn, L2P / 16), 256, 0, stream>>>(q2b, q1b, mask1, P1, L1P, L2P, L1);
  // weighted2: Q=q1, K=q2, mask over n+L2 columns
  attn_probs<<<dim3(Bn, L1P / 16), 256, 0, stream>>>(q1b, q2b, mask2, P2, L2P, L1P, L2);

  // out1 rows 3..32 live in m-tiles 0..2; out2 rows 3..198 in m-tiles 0..12
  pv_gemm<<<dim3(Bn, 3, D1 / 256), 256, 0, stream>>>(P1, v1T, out1, L1P, D1, L2, L2P);
  pv_gemm<<<dim3(Bn, 13, D2 / 256), 256, 0, stream>>>(P2, v2T, out2, L2P, D2, L1, L1P);
}
